// GNNStack_38878043964109
// MI455X (gfx1250) — matrix operand order repaired and checked by reading, not compiled, not run on hardware
//
#include <hip/hip_runtime.h>
#include <hip/hip_bf16.h>

typedef __attribute__((ext_vector_type(16))) _Float16 v16h;
typedef __attribute__((ext_vector_type(8)))  float    v8f;

// ---------------------------------------------------------------------------
// Degree / normalization:  dinv[i] = rsqrt(1 + indegree(i))
// ---------------------------------------------------------------------------
__global__ void deg_init_kernel(float* __restrict__ deg, int n) {
    int i = blockIdx.x * blockDim.x + threadIdx.x;
    if (i < n) deg[i] = 1.0f;   // self loop
}

__global__ void deg_count_kernel(const int* __restrict__ dst, float* __restrict__ deg, int e) {
    int i = blockIdx.x * blockDim.x + threadIdx.x;
    if (i < e) unsafeAtomicAdd(&deg[dst[i]], 1.0f);
}

__global__ void deg_finalize_kernel(float* __restrict__ deg, int n) {
    int i = blockIdx.x * blockDim.x + threadIdx.x;
    if (i < n) deg[i] = rsqrtf(deg[i]);
}

// ---------------------------------------------------------------------------
// WMMA GEMM with fused GCN normalization.
//   pre(A)[r][k]  = [PRER] relu( [PRESCALE] A[r][k]*dinv[r] + [PREB] abias[k] )
//   val[r][c]     = pre(A) @ W  (+ [POSTB] outb[c]) ([POSTR] relu)
//   store:  Out[r][c]  = val * ([POSTSCALE] dinv[r])
//           [DUAL] Out2[r][c] = same   (scatter accumulator init = self-loop term)
// One wave -> one 16-row tile x NT column tiles, K=64 = 2 WMMA k-steps.
// NOTE: A may alias Out2 (layers 2/3 read u from and re-init the same buffer);
// per-tile rows are read before written by the owning wave, so no __restrict__
// on those two pointers.
// ---------------------------------------------------------------------------
template<int NT, bool PRESCALE, bool PREB, bool PRER,
         bool POSTB, bool POSTR, bool POSTSCALE, bool DUAL>
__global__ __launch_bounds__(256)
void gemm_k64_kernel(const float* A, const float* __restrict__ Wg,
                     const float* __restrict__ abias, const float* __restrict__ outb,
                     const float* __restrict__ dinv,
                     float* __restrict__ Out, float* Out2, int M)
{
    constexpr int NC = NT * 16;
    __shared__ _Float16 Wl[64 * NC];
    for (int i = threadIdx.x; i < 64 * NC; i += blockDim.x)
        Wl[i] = (_Float16)Wg[i];
    __syncthreads();

    const int lane = threadIdx.x & 31;
    const int sub  = lane & 15;
    const int hi   = lane >> 4;            // 0: lanes 0-15, 1: lanes 16-31

    // B fragments (32x16 f16): elements 0-7 = K kb..kb+7, elements 8-15 = K kb+16..kb+23 (kb includes hi*8)
    v16h bf[NT][2];
    #pragma unroll
    for (int ct = 0; ct < NT; ++ct) {
        #pragma unroll
        for (int ks = 0; ks < 2; ++ks) {
            const int kb  = ks * 32 + hi * 8;
            const int col = ct * 16 + sub;
            v16h b;
            #pragma unroll
            for (int j = 0; j < 16; ++j)
                b[j] = Wl[(kb + (j < 8 ? j : j + 8)) * NC + col];
            bf[ct][ks] = b;
        }
    }

    // Hoist pre-bias into registers (invariant across row tiles).
    float ab[2][16];
    if (PREB) {
        #pragma unroll
        for (int ks = 0; ks < 2; ++ks) {
            const int kb = ks * 32 + hi * 8;
            #pragma unroll
            for (int j = 0; j < 8; ++j) {
                ab[ks][j]     = abias[kb + j];
                ab[ks][8 + j] = abias[kb + 16 + j];
            }
        }
    }
    float obv[NT];
    if (POSTB) {
        #pragma unroll
        for (int ct = 0; ct < NT; ++ct) obv[ct] = outb[ct * 16 + sub];
    }

    const int nTiles = (M + 15) >> 4;
    const int waveId = blockIdx.x * (blockDim.x >> 5) + (threadIdx.x >> 5);
    const int nWaves = gridDim.x * (blockDim.x >> 5);

    for (int t = waveId; t < nTiles; t += nWaves) {
        const int r0 = t << 4;
        int row = r0 + sub;
        if (row >= M) row = M - 1;                   // clamp (M % 16 == 0 normally)
        const float4* ar4 = (const float4*)(A + (size_t)row * 64);

        // One dinv load per lane; store-phase row scales come from lane shuffles.
        float d = 1.0f;
        if (PRESCALE || POSTSCALE) d = dinv[row];

        v8f acc[NT];
        #pragma unroll
        for (int ct = 0; ct < NT; ++ct) acc[ct] = (v8f){0,0,0,0,0,0,0,0};

        #pragma unroll
        for (int ks = 0; ks < 2; ++ks) {
            // A fragment (16x32 f16): halves 0-7 = K kb..kb+7, halves 8-15 = K kb+16..kb+23
            const int kq = ks * 8 + hi * 2;          // (ks*32 + hi*8) / 4
            const float4 p0 = ar4[kq + 0];
            const float4 p1 = ar4[kq + 1];
            const float4 p2 = ar4[kq + 4];
            const float4 p3 = ar4[kq + 5];
            const float tv[16] = { p0.x, p0.y, p0.z, p0.w, p1.x, p1.y, p1.z, p1.w,
                                   p2.x, p2.y, p2.z, p2.w, p3.x, p3.y, p3.z, p3.w };
            v16h a;
            #pragma unroll
            for (int j = 0; j < 16; ++j) {
                float v = tv[j];
                if (PRESCALE) v *= d;
                if (PREB)     v += ab[ks][j];
                if (PRER)     v = v > 0.f ? v : 0.f;
                a[j] = (_Float16)v;
            }
            #pragma unroll
            for (int ct = 0; ct < NT; ++ct)
                acc[ct] = __builtin_amdgcn_wmma_f32_16x16x32_f16(
                    false, a, false, bf[ct][ks], (short)0, acc[ct], false, false);
        }

        // Row scales for the 8 output rows this lane stores: lane (hi*8+i) holds
        // dinv[r0 + hi*8 + i] in d  ->  wave shuffle instead of 8 clamped loads.
        float rs[8];
        if (POSTSCALE) {
            #pragma unroll
            for (int i = 0; i < 8; ++i)
                rs[i] = __shfl(d, hi * 8 + i, 32);
        }

        // C/D layout: VGPR i -> M = i + hi*8, N = sub.
        // Hoisted wave-uniform full/tail branch: fast path is 8 stores per column
        // tile off one base pointer with immediate offsets (i*NC*4 <= 1792B).
        if (r0 + 16 <= M) {
            #pragma unroll
            for (int ct = 0; ct < NT; ++ct) {
                const size_t base = (size_t)(r0 + hi * 8) * NC + ct * 16 + sub;
                float* __restrict__ p1 = Out + base;
                float*              p2 = DUAL ? (Out2 + base) : nullptr;
                #pragma unroll
                for (int i = 0; i < 8; ++i) {
                    float v = acc[ct][i];
                    if (POSTB) v += obv[ct];
                    if (POSTR) v = v > 0.f ? v : 0.f;
                    if (POSTSCALE) v *= rs[i];
                    p1[i * NC] = v;
                    if (DUAL) p2[i * NC] = v;
                }
            }
        } else {
            #pragma unroll
            for (int ct = 0; ct < NT; ++ct) {
                const int col = ct * 16 + sub;
                #pragma unroll
                for (int i = 0; i < 8; ++i) {
                    const int rr = r0 + i + hi * 8;
                    if (rr < M) {
                        float v = acc[ct][i];
                        if (POSTB) v += obv[ct];
                        if (POSTR) v = v > 0.f ? v : 0.f;
                        if (POSTSCALE) v *= rs[i];
                        const size_t off = (size_t)rr * NC + col;
                        Out[off] = v;
                        if (DUAL) Out2[off] = v;
                    }
                }
            }
        }
    }
}

// ---------------------------------------------------------------------------
// Edge scatter: agg[dst,:] += h'[src,:]   (h' pre-scaled by dinv; pure atomics)
// ---------------------------------------------------------------------------
__global__ void scatter_kernel(const float* __restrict__ h,
                               const int* __restrict__ src, const int* __restrict__ dst,
                               float* __restrict__ agg, long long ne)
{
    long long idx = (long long)blockIdx.x * blockDim.x + threadIdx.x;
    if (idx >= ne * 16) return;
    const long long e = idx >> 4;
    const int q = (int)(idx & 15);
    const int s = src[e];
    const int d = dst[e];
    const float4 v = ((const float4*)(h + (size_t)s * 64))[q];
    float* ap = agg + (size_t)d * 64 + q * 4;
    unsafeAtomicAdd(ap + 0, v.x);
    unsafeAtomicAdd(ap + 1, v.y);
    unsafeAtomicAdd(ap + 2, v.z);
    unsafeAtomicAdd(ap + 3, v.w);
}

// ---------------------------------------------------------------------------
// Head tail: z2 = z1[32] @ Wp2[32x40] + bp2; out = log_softmax(z2)
// ---------------------------------------------------------------------------
__global__ __launch_bounds__(256)
void head2_kernel(const float* __restrict__ z1, const float* __restrict__ Wp2,
                  const float* __restrict__ bp2, float* __restrict__ out, int n)
{
    __shared__ float Wl[32 * 40];
    __shared__ float bl[40];
    for (int i = threadIdx.x; i < 32 * 40; i += blockDim.x) Wl[i] = Wp2[i];
    for (int i = threadIdx.x; i < 40; i += blockDim.x)      bl[i] = bp2[i];
    __syncthreads();

    const int node = blockIdx.x * blockDim.x + threadIdx.x;
    if (node >= n) return;

    float zin[32];
    const float4* zp = (const float4*)(z1 + (size_t)node * 32);
    #pragma unroll
    for (int q = 0; q < 8; ++q) {
        float4 v = zp[q];
        zin[4*q+0] = v.x; zin[4*q+1] = v.y; zin[4*q+2] = v.z; zin[4*q+3] = v.w;
    }

    float z[40];
    #pragma unroll
    for (int j = 0; j < 40; ++j) {
        float s = bl[j];
        #pragma unroll
        for (int k = 0; k < 32; ++k) s += zin[k] * Wl[k * 40 + j];
        z[j] = s;
    }

    float m = z[0];
    #pragma unroll
    for (int j = 1; j < 40; ++j) m = z[j] > m ? z[j] : m;
    float ssum = 0.f;
    #pragma unroll
    for (int j = 0; j < 40; ++j) ssum += __expf(z[j] - m);
    const float lse = m + __logf(ssum);

    float* op = out + (size_t)node * 40;
    #pragma unroll
    for (int j = 0; j < 40; ++j) op[j] = z[j] - lse;
}

// ---------------------------------------------------------------------------
// Host launcher
// ---------------------------------------------------------------------------
static inline size_t align256(size_t x) { return (x + 255) & ~(size_t)255; }

extern "C" void kernel_launch(void* const* d_in, const int* in_sizes, int n_in,
                              void* d_out, int out_size, void* d_ws, size_t ws_size,
                              hipStream_t stream)
{
    const float* x   = (const float*)d_in[0];
    const int*   ei  = (const int*)d_in[1];
    // d_in[2] = batch (unused for node task)
    const float* W0  = (const float*)d_in[3];  const float* b0  = (const float*)d_in[4];
    const float* W1  = (const float*)d_in[5];  const float* b1  = (const float*)d_in[6];
    const float* W2  = (const float*)d_in[7];  const float* b2  = (const float*)d_in[8];
    const float* Wp1 = (const float*)d_in[9];  const float* bp1 = (const float*)d_in[10];
    const float* Wp2 = (const float*)d_in[11]; const float* bp2 = (const float*)d_in[12];

    const int n = in_sizes[0] / 64;       // nodes
    const int e = in_sizes[1] / 2;        // edges
    const int* src = ei;
    const int* dst = ei + e;

    // workspace layout: dinv[n] | bufA[n*64] | bufB[n*64]
    char* ws = (char*)d_ws;
    float* dinv = (float*)ws;
    float* bufA = (float*)(ws + align256((size_t)n * sizeof(float)));
    float* bufB = (float*)((char*)bufA + align256((size_t)n * 64 * sizeof(float)));
    float* outp = (float*)d_out;

    const int B = 256;
    const int gN  = (n + B - 1) / B;
    const int gE  = (e + B - 1) / B;
    const long long eq = (long long)e * 16;
    const int gEQ = (int)((eq + B - 1) / B);
    const int nTiles = (n + 15) >> 4;
    const int gG  = (nTiles + 15) / 16;   // 8 waves/block, ~2 row tiles/wave

    // dinv = rsqrt(1 + indegree)
    deg_init_kernel<<<gN, B, 0, stream>>>(dinv, n);
    deg_count_kernel<<<gE, B, 0, stream>>>(dst, dinv, e);
    deg_finalize_kernel<<<gN, B, 0, stream>>>(dinv, n);

    // Layer 1: h' = (x @ W0) * dinv[row]  -> bufA (gather src) and bufB (accumulator)
    gemm_k64_kernel<4, false, false, false, false, false, true, true>
        <<<gG, B, 0, stream>>>(x, W0, nullptr, nullptr, dinv, bufA, bufB, n);
    scatter_kernel<<<gEQ, B, 0, stream>>>(bufA, src, dst, bufB, (long long)e);

    // Layer 2: h' = (relu(u*dinv + b0) @ W1) * dinv  -> bufA, bufB
    gemm_k64_kernel<4, true, true, true, false, false, true, true>
        <<<gG, B, 0, stream>>>(bufB, W1, b0, nullptr, dinv, bufA, bufB, n);
    scatter_kernel<<<gEQ, B, 0, stream>>>(bufA, src, dst, bufB, (long long)e);

    // Layer 3
    gemm_k64_kernel<4, true, true, true, false, false, true, true>
        <<<gG, B, 0, stream>>>(bufB, W2, b1, nullptr, dinv, bufA, bufB, n);
    scatter_kernel<<<gEQ, B, 0, stream>>>(bufA, src, dst, bufB, (long long)e);

    // Head: z1 = relu( relu(u*dinv + b2) @ Wp1 + bp1 )   [N x 32] -> bufA
    gemm_k64_kernel<2, true, true, true, true, true, false, false>
        <<<gG, B, 0, stream>>>(bufB, Wp1, b2, bp1, dinv, bufA, nullptr, n);

    // z2 = z1 @ Wp2 + bp2 ; log_softmax
    head2_kernel<<<gN, B, 0, stream>>>(bufA, Wp2, bp2, outp, n);
}
